// ADDSERQDiT_35459249996354
// MI455X (gfx1250) — compile-verified, matrix operand order checked by reading, not run on hardware
//
#include <hip/hip_runtime.h>
#include <math.h>

// ---------- types ----------
typedef __attribute__((ext_vector_type(8)))  float  v8f;
typedef __attribute__((ext_vector_type(16))) __bf16 bf16x16;
typedef __attribute__((ext_vector_type(4)))  __bf16 bf16x4;

#define SSC 0.70710678118654752f   // 2^-0.5
#define HID 512
#define HEADS 8
#define HD 64

static __device__ inline __bf16 f2bf(float f) { return (__bf16)f; }  // native v_cvt, RNE

static __device__ inline bf16x4 pack4(float a, float b, float c, float d) {
  bf16x4 r; r[0] = (__bf16)a; r[1] = (__bf16)b; r[2] = (__bf16)c; r[3] = (__bf16)d;
  return r;
}

// Load a 16x32 bf16 WMMA fragment (A layout; B uses same layout on B^T tile).
// p points at LDS row (m or n) base + kl; row stride padded so addresses are
// 16B-aligned -> compiler merges into ds_load_b128 pairs.
static __device__ inline bf16x16 load_frag(const __bf16* p) {
  bf16x4 a0 = *(const bf16x4*)(p);
  bf16x4 a1 = *(const bf16x4*)(p + 4);
  bf16x4 a2 = *(const bf16x4*)(p + 16);
  bf16x4 a3 = *(const bf16x4*)(p + 20);
  bf16x16 r;
  for (int i = 0; i < 4; ++i) { r[i] = a0[i]; r[i+4] = a1[i]; r[i+8] = a2[i]; r[i+12] = a3[i]; }
  return r;
}

// =====================================================================
// GEMM: C[M,N] = act(A[M,Kd] @ W[Kd,N] + bias), fp32 in, bf16 WMMA, fp32 acc.
// Tile 128x128x32, 256 threads = 8 waves (2x4), each wave 64x32 (4x2 wmma tiles).
// act: 0=none, 1=gelu(tanh). outmode: 0 -> C[m*N+n]; 1 -> DiT final scatter
// (m = k*1024+l within chunk bidx -> C[((bidx*64+n)*4+k)*1024+l]).
// M must be a multiple of 128, Kd a multiple of 32, N a multiple of 32.
// =====================================================================
__global__ __launch_bounds__(256) void gemm_wmma(
    const float* __restrict__ A, const float* __restrict__ W,
    const float* __restrict__ bias, float* __restrict__ C,
    int M, int N, int Kd, int act, int outmode, int bidx)
{
  __shared__ __bf16 As[128 * 40];
  __shared__ __bf16 Bs[128 * 40];
  const int tid  = threadIdx.x;
  const int lane = tid & 31, wid = tid >> 5;
  const int wr = wid >> 2, wc = wid & 3;
  const int rl = lane & 15, kl = (lane >> 4) << 3, hi = lane >> 4;
  const int m0 = blockIdx.y * 128, n0 = blockIdx.x * 128;

  v8f acc[4][2];
  for (int f = 0; f < 4; ++f) for (int g = 0; g < 2; ++g)
    for (int v = 0; v < 8; ++v) acc[f][g][v] = 0.f;

  const int arow = tid >> 3;           // 0..31
  const int acol = (tid & 7) * 4;      // 0..28

  for (int k0 = 0; k0 < Kd; k0 += 32) {
    // ---- issue all global tile loads first (amortize LOADcnt waits) ----
    float va[4][4], vb[4][4];
    for (int p = 0; p < 4; ++p) {
      int r = p * 32 + arow;
      float4 v4 = *(const float4*)(A + (size_t)(m0 + r) * Kd + k0 + acol);
      va[p][0] = v4.x; va[p][1] = v4.y; va[p][2] = v4.z; va[p][3] = v4.w;
    }
    for (int p = 0; p < 4; ++p) {
      int nn = p * 32 + (tid & 7) * 4;
      // Uniform slab guard (N % 32 == 0): whole 32-col slab in range or zero.
      if (n0 + p * 32 < N) {
        float4 v4 = *(const float4*)(W + (size_t)(k0 + arow) * N + n0 + nn);
        vb[p][0] = v4.x; vb[p][1] = v4.y; vb[p][2] = v4.z; vb[p][3] = v4.w;
      } else {
        vb[p][0] = 0.f; vb[p][1] = 0.f; vb[p][2] = 0.f; vb[p][3] = 0.f;
      }
    }
    // ---- convert + store to LDS ----
    for (int p = 0; p < 4; ++p) {
      int r = p * 32 + arow;
      *(bf16x4*)&As[r * 40 + acol] = pack4(va[p][0], va[p][1], va[p][2], va[p][3]);
    }
    for (int p = 0; p < 4; ++p) {
      int nn = p * 32 + (tid & 7) * 4;
      for (int j = 0; j < 4; ++j) Bs[(nn + j) * 40 + arow] = f2bf(vb[p][j]);
    }
    if (k0 + 32 < Kd) __builtin_prefetch(A + (size_t)(m0 + arow) * Kd + k0 + 32, 0, 1);
    __syncthreads();

    bf16x16 af[4], bfr[2];
    for (int f = 0; f < 4; ++f) af[f]  = load_frag(&As[(wr * 64 + f * 16 + rl) * 40 + kl]);
    for (int g = 0; g < 2; ++g) bfr[g] = load_frag(&Bs[(wc * 32 + g * 16 + rl) * 40 + kl]);
    for (int f = 0; f < 4; ++f)
      for (int g = 0; g < 2; ++g)
        acc[f][g] = __builtin_amdgcn_wmma_f32_16x16x32_bf16(
            false, af[f], false, bfr[g], (short)0, acc[f][g], false, false);
    __syncthreads();
  }

  for (int f = 0; f < 4; ++f)
    for (int g = 0; g < 2; ++g) {
      int n = n0 + wc * 32 + g * 16 + rl;
      if (n >= N) continue;
      float bv = bias ? bias[n] : 0.f;
      for (int v = 0; v < 8; ++v) {
        int m = m0 + wr * 64 + f * 16 + v + hi * 8;
        float val = acc[f][g][v] + bv;
        if (act == 1) {
          float x3 = val * val * val;
          val = 0.5f * val * (1.f + tanhf(0.7978845608f * (val + 0.044715f * x3)));
        }
        if (outmode == 0) C[(size_t)m * N + n] = val;
        else {
          int kk = m >> 10, ll = m & 1023;
          C[(((size_t)bidx * 64 + n) * 4 + kk) * 1024 + ll] = val;
        }
      }
    }
}

// =====================================================================
// Flash attention (time stage): grid (L/128, HEADS, B), 256 threads.
// Q/K/V: [B,H,L,64] fp32. O: [B*L, 512] (cols h*64+d).
// Tiles: 128 queries x 64 keys, online softmax, WMMA bf16.
// =====================================================================
__global__ __launch_bounds__(256) void flash_attn(
    const float* __restrict__ Q, const float* __restrict__ Kb,
    const float* __restrict__ Vb, float* __restrict__ O)
{
  __shared__ __bf16 Qs[128 * 72];
  __shared__ __bf16 Ks[64 * 72];
  __shared__ __bf16 Vt[64 * 72];
  __shared__ float  Ss[128 * 72];
  __shared__ __bf16 Ps[128 * 72];
  __shared__ float  rowm[128], rowsum[128], rowa[128];

  const int tid = threadIdx.x, lane = tid & 31, wid = tid >> 5;
  const int wr = wid >> 2, wc = wid & 3;
  const int rl = lane & 15, kl = (lane >> 4) << 3, hi = lane >> 4;
  const int q0 = blockIdx.x * 128;
  const int h = blockIdx.y, b = blockIdx.z;
  const size_t base = ((size_t)(b * HEADS + h)) * 1024 * HD;

  { // Q tile (vectorized cvt + ds_store_b64)
    int r = tid >> 1, d0 = (tid & 1) * 32;
    const float* src = Q + base + (size_t)(q0 + r) * HD + d0;
    for (int i = 0; i < 32; i += 4) {
      float4 v4 = *(const float4*)(src + i);
      *(bf16x4*)&Qs[r * 72 + d0 + i] = pack4(v4.x, v4.y, v4.z, v4.w);
    }
  }
  if (tid < 128) { rowm[tid] = -1e30f; rowsum[tid] = 0.f; }

  v8f oacc[4];
  for (int f = 0; f < 4; ++f) for (int v = 0; v < 8; ++v) oacc[f][v] = 0.f;

  for (int kv0 = 0; kv0 < 1024; kv0 += 64) {
    __syncthreads();   // protect Ks/Vt/Ps from previous iteration
    { // K,V tiles (V transposed into Vt[d][kk])
      int r = tid >> 2, d0 = (tid & 3) * 16;
      const float* ks = Kb + base + (size_t)(kv0 + r) * HD + d0;
      const float* vs = Vb + base + (size_t)(kv0 + r) * HD + d0;
      for (int i = 0; i < 16; i += 4) {
        float4 kv = *(const float4*)(ks + i);
        float4 vv = *(const float4*)(vs + i);
        *(bf16x4*)&Ks[r * 72 + d0 + i] = pack4(kv.x, kv.y, kv.z, kv.w);
        Vt[(d0 + i + 0) * 72 + r] = f2bf(vv.x);
        Vt[(d0 + i + 1) * 72 + r] = f2bf(vv.y);
        Vt[(d0 + i + 2) * 72 + r] = f2bf(vv.z);
        Vt[(d0 + i + 3) * 72 + r] = f2bf(vv.w);
      }
    }
    __syncthreads();

    // S = (Q K^T) * hd^-0.5   [128 q x 64 kk]; wave: 4x1 tiles, d-loop 2
    v8f sacc[4];
    for (int f = 0; f < 4; ++f) for (int v = 0; v < 8; ++v) sacc[f][v] = 0.f;
    for (int kt = 0; kt < 64; kt += 32) {
      bf16x16 bfr = load_frag(&Ks[(wc * 16 + rl) * 72 + kt + kl]);
      for (int f = 0; f < 4; ++f) {
        bf16x16 afr = load_frag(&Qs[(wr * 64 + f * 16 + rl) * 72 + kt + kl]);
        sacc[f] = __builtin_amdgcn_wmma_f32_16x16x32_bf16(
            false, afr, false, bfr, (short)0, sacc[f], false, false);
      }
    }
    for (int f = 0; f < 4; ++f) {
      int kkcol = wc * 16 + rl;
      for (int v = 0; v < 8; ++v) {
        int qr = wr * 64 + f * 16 + v + hi * 8;
        Ss[qr * 72 + kkcol] = sacc[f][v] * 0.125f;
      }
    }
    __syncthreads();

    if (tid < 128) { // online softmax, one row per thread
      float mo = rowm[tid], mx = mo;
      for (int j = 0; j < 64; ++j) mx = fmaxf(mx, Ss[tid * 72 + j]);
      float alpha = __expf(mo - mx);
      float s = 0.f;
      for (int j = 0; j < 64; ++j) {
        float p = __expf(Ss[tid * 72 + j] - mx);
        s += p;
        Ps[tid * 72 + j] = f2bf(p);
      }
      rowm[tid] = mx;
      rowsum[tid] = rowsum[tid] * alpha + s;
      rowa[tid] = alpha;
    }
    __syncthreads();

    // O = O*alpha + P V
    for (int f = 0; f < 4; ++f)
      for (int v = 0; v < 8; ++v) {
        int qr = wr * 64 + f * 16 + v + hi * 8;
        oacc[f][v] *= rowa[qr];
      }
    for (int kt = 0; kt < 64; kt += 32) {
      bf16x16 bfr = load_frag(&Vt[(wc * 16 + rl) * 72 + kt + kl]);
      for (int f = 0; f < 4; ++f) {
        bf16x16 afr = load_frag(&Ps[(wr * 64 + f * 16 + rl) * 72 + kt + kl]);
        oacc[f] = __builtin_amdgcn_wmma_f32_16x16x32_bf16(
            false, afr, false, bfr, (short)0, oacc[f], false, false);
      }
    }
  }
  __syncthreads();
  for (int f = 0; f < 4; ++f) {
    int d = wc * 16 + rl;
    for (int v = 0; v < 8; ++v) {
      int qr = wr * 64 + f * 16 + v + hi * 8;
      O[(size_t)(b * 1024 + q0 + qr) * HID + h * HD + d] = oacc[f][v] / rowsum[qr];
    }
  }
}

// =====================================================================
// Dep-stage attention: seq len 4, rope over position k. One wave per (l,h).
// qkv: chunk [4096,1536] rows m = l*4+k. O: [4096,512].
// =====================================================================
__global__ __launch_bounds__(256) void dep_attn(
    const float* __restrict__ qkv, float* __restrict__ O)
{
  int l = blockIdx.x;
  int h = threadIdx.x >> 5;
  int j = threadIdx.x & 31;              // d-pair index
  float q[4][2], k[4][2], v[4][2];
  float omega = powf(10000.f, -(float)j / 32.f);
  for (int i = 0; i < 4; ++i) {
    size_t row = (size_t)(l * 4 + i) * 1536;
    size_t o = row + h * 64 + 2 * j;
    float q0 = qkv[o], q1 = qkv[o + 1];
    float k0 = qkv[o + 512], k1 = qkv[o + 513];
    v[i][0] = qkv[o + 1024]; v[i][1] = qkv[o + 1025];
    float ang = (float)i * omega, cc = cosf(ang), ss = sinf(ang);
    q[i][0] = q0 * cc - q1 * ss; q[i][1] = q1 * cc + q0 * ss;
    k[i][0] = k0 * cc - k1 * ss; k[i][1] = k1 * cc + k0 * ss;
  }
  float s[4][4];
  for (int i = 0; i < 4; ++i)
    for (int jj = 0; jj < 4; ++jj) {
      float p = q[i][0] * k[jj][0] + q[i][1] * k[jj][1];
      for (int off = 16; off > 0; off >>= 1) p += __shfl_xor(p, off, 32);
      s[i][jj] = p * 0.125f;
    }
  for (int i = 0; i < 4; ++i) {
    float mx = fmaxf(fmaxf(s[i][0], s[i][1]), fmaxf(s[i][2], s[i][3]));
    float e[4], sum = 0.f;
    for (int jj = 0; jj < 4; ++jj) { e[jj] = __expf(s[i][jj] - mx); sum += e[jj]; }
    float o0 = 0.f, o1 = 0.f;
    for (int jj = 0; jj < 4; ++jj) { float p = e[jj] / sum; o0 += p * v[jj][0]; o1 += p * v[jj][1]; }
    size_t od = (size_t)(l * 4 + i) * HID + h * 64 + 2 * j;
    O[od] = o0; O[od + 1] = o1;
  }
}

// ---------- elementwise / small kernels ----------
__global__ void k_rope(const float* __restrict__ qkv, float* __restrict__ Qr,
                       float* __restrict__ Kr, float* __restrict__ Vr)
{
  int t = blockIdx.x * 256 + threadIdx.x;      // 4096*8*32 total
  int j = t & 31, h = (t >> 5) & 7, m = t >> 8;
  int b = m >> 10, l = m & 1023;
  size_t base = (size_t)m * 1536 + h * 64 + 2 * j;
  float q0 = qkv[base], q1 = qkv[base + 1];
  float k0 = qkv[base + 512], k1 = qkv[base + 513];
  float v0 = qkv[base + 1024], v1 = qkv[base + 1025];
  float omega = powf(10000.f, -(float)j / 32.f);
  float ang = (float)l * omega, cc = cosf(ang), ss = sinf(ang);
  size_t o = ((size_t)(b * 8 + h) * 1024 + l) * 64 + 2 * j;
  Qr[o] = q0 * cc - q1 * ss; Qr[o + 1] = q1 * cc + q0 * ss;
  Kr[o] = k0 * cc - k1 * ss; Kr[o + 1] = k1 * cc + k0 * ss;
  Vr[o] = v0; Vr[o + 1] = v1;
}

__global__ void k_ksum(const float* __restrict__ xh, float* __restrict__ out)
{
  int t = blockIdx.x * 256 + threadIdx.x;      // 4096*512
  int m = t >> 9, j = t & 511, b = m >> 10, l = m & 1023;
  float s = 0.f;
  for (int kk = 0; kk < 4; ++kk) s += xh[((size_t)(b * 4 + kk) * 1024 + l) * HID + j];
  out[t] = s * 0.5f;
}

__global__ void k_cond_time(const float* __restrict__ ch, const float* __restrict__ te,
                            float* __restrict__ out)
{
  int t = blockIdx.x * 256 + threadIdx.x;      // 4096*512
  int m = t >> 9, j = t & 511, b = m >> 10, l = m & 1023;
  float s = 0.f;
  for (int kk = 0; kk < 4; ++kk) s += ch[((size_t)(b * 4 + kk) * 1024 + l) * HID + j];
  float val = (s * 0.5f + te[b * HID + j]) * SSC;
  out[t] = val / (1.f + __expf(-val));         // silu
}

__global__ void k_cond_dep(const float* __restrict__ ch, const float* __restrict__ te,
                           float* __restrict__ out, int b, int order)
{
  int t = blockIdx.x * 256 + threadIdx.x;      // 4096*512
  int r = t >> 9, j = t & 511, l, kk;
  if (order == 0) { l = r >> 2; kk = r & 3; } else { kk = r >> 10; l = r & 1023; }
  float val = (ch[((size_t)(b * 4 + kk) * 1024 + l) * HID + j] + te[b * HID + j]) * SSC;
  out[t] = val / (1.f + __expf(-val));
}

__global__ void k_build_dep(const float* __restrict__ xh, const float* __restrict__ hh,
                            float* __restrict__ depX)
{
  int t = blockIdx.x * 256 + threadIdx.x;      // 16384*512
  int m = t >> 9, j = t & 511;
  int b = m >> 12, rem = m & 4095, l = rem >> 2, kk = rem & 3;
  depX[t] = (xh[((size_t)(b * 4 + kk) * 1024 + l) * HID + j] +
             hh[((size_t)b * 1024 + l) * HID + j]) * SSC;
}

__global__ void k_gather_final(const float* __restrict__ depX, float* __restrict__ out, int b)
{
  int t = blockIdx.x * 256 + threadIdx.x;      // 4096*512
  int r = t >> 9, j = t & 511, kk = r >> 10, l = r & 1023;
  out[t] = depX[((size_t)b * 4096 + l * 4 + kk) * HID + j];
}

__global__ void k_gather_in(const float* __restrict__ x, float* __restrict__ out)
{
  int t = blockIdx.x * 256 + threadIdx.x;      // 16384*64
  int m = t >> 6, cin = t & 63;
  int b = m >> 12, rem = m & 4095, kk = rem >> 10, l = rem & 1023;
  out[t] = x[(((size_t)b * 64 + cin) * 4 + kk) * 1024 + l];
}

__global__ __launch_bounds__(256) void k_lnmod(
    const float* __restrict__ X, const float* __restrict__ mods, int mstride,
    int sh_off, int sc_off, float* __restrict__ out)
{
  __shared__ float r1[256], r2[256];
  int row = blockIdx.x, t = threadIdx.x;
  float a = X[(size_t)row * HID + t], b2 = X[(size_t)row * HID + 256 + t];
  r1[t] = a + b2; r2[t] = a * a + b2 * b2;
  __syncthreads();
  for (int s = 128; s > 0; s >>= 1) {
    if (t < s) { r1[t] += r1[t + s]; r2[t] += r2[t + s]; }
    __syncthreads();
  }
  float mean = r1[0] / 512.f;
  float var = r2[0] / 512.f - mean * mean;
  float rstd = rsqrtf(var + 1e-6f);
  const float* mrow = mods + (size_t)row * mstride;
  out[(size_t)row * HID + t] =
      ((a - mean) * rstd * (1.f + mrow[sc_off + t]) + mrow[sh_off + t]) * SSC;
  out[(size_t)row * HID + 256 + t] =
      ((b2 - mean) * rstd * (1.f + mrow[sc_off + 256 + t]) + mrow[sh_off + 256 + t]) * SSC;
}

__global__ void k_resid(float* __restrict__ X, const float* __restrict__ mods,
                        int goff, const float* __restrict__ Y)
{
  int t = blockIdx.x * 256 + threadIdx.x;      // 4096*512
  int m = t >> 9, j = t & 511;
  X[t] = (X[t] + mods[(size_t)m * 3072 + goff + j] * Y[t]) * SSC;
}

__global__ void k_tfeat(const float* __restrict__ tt, const float* __restrict__ freqs,
                        const float* __restrict__ phases, float* __restrict__ f)
{
  int b = blockIdx.x, e = threadIdx.x;         // grid 4 x 256
  f[b * 256 + e] = 1.41421356237f * cosf(tt[b] * freqs[e] + phases[e]);
}

__global__ void k_dense_small(const float* __restrict__ in, const float* __restrict__ W,
                              const float* __restrict__ bias, float* __restrict__ out,
                              int Bn, int din, int dout, int act)
{
  int t = blockIdx.x * 256 + threadIdx.x;
  if (t >= Bn * dout) return;
  int b = t / dout, jj = t % dout;
  float s = bias[jj];
  for (int i = 0; i < din; ++i) s += in[b * din + i] * W[(size_t)i * dout + jj];
  if (act == 1) s = s / (1.f + __expf(-s));
  out[t] = s;
}

// =====================================================================
// Host orchestration
// =====================================================================
struct BlockP {
  const float *qkv_w, *qkv_b, *proj_w, *proj_b, *m1w, *m1b, *m2w, *m2b, *aw, *ab;
};

extern "C" void kernel_launch(void* const* d_in, const int* in_sizes, int n_in,
                              void* d_out, int out_size, void* d_ws, size_t ws_size,
                              hipStream_t stream) {
  (void)in_sizes; (void)n_in; (void)out_size; (void)ws_size;
  const float* x  = (const float*)d_in[0];
  const float* c  = (const float*)d_in[1];
  const float* tt = (const float*)d_in[2];
  const float* in_x_w = (const float*)d_in[3];
  const float* in_x_b = (const float*)d_in[4];
  const float* in_c_w = (const float*)d_in[5];
  const float* in_c_b = (const float*)d_in[6];
  const float* t_w1 = (const float*)d_in[7];
  const float* t_b1 = (const float*)d_in[8];
  const float* t_w2 = (const float*)d_in[9];
  const float* t_b2 = (const float*)d_in[10];
  const float* out_ada_w = (const float*)d_in[11];
  const float* out_ada_b = (const float*)d_in[12];
  const float* out_w = (const float*)d_in[13];
  const float* out_b = (const float*)d_in[14];
  const float* freqs = (const float*)d_in[15];
  const float* phases = (const float*)d_in[16];
  auto getBlock = [&](int base) {
    BlockP P;
    P.qkv_w = (const float*)d_in[base + 0]; P.qkv_b = (const float*)d_in[base + 1];
    P.proj_w = (const float*)d_in[base + 2]; P.proj_b = (const float*)d_in[base + 3];
    P.m1w = (const float*)d_in[base + 4]; P.m1b = (const float*)d_in[base + 5];
    P.m2w = (const float*)d_in[base + 6]; P.m2b = (const float*)d_in[base + 7];
    P.aw = (const float*)d_in[base + 8]; P.ab = (const float*)d_in[base + 9];
    return P;
  };
  BlockP tb[4], db[4];
  for (int i = 0; i < 4; ++i) tb[i] = getBlock(17 + i * 10);
  for (int i = 0; i < 4; ++i) db[i] = getBlock(57 + i * 10);

  // ---- workspace carve (floats) ----
  float* ws = (float*)d_ws;
  size_t off = 0;
  auto alloc = [&](size_t n) { float* p = ws + off; off += n; return p; };
  float* xh    = alloc((size_t)16384 * 512);
  float* ch    = alloc((size_t)16384 * 512);
  float* depX  = alloc((size_t)16384 * 512);
  float* x_cur = alloc((size_t)4096 * 512);
  float* sact  = alloc((size_t)4096 * 512);
  float* mods  = alloc((size_t)4096 * 3072);
  float* tok   = alloc((size_t)4096 * 512);
  float* qkv   = alloc((size_t)4096 * 1536);
  float* Qr    = alloc((size_t)4096 * 512);
  float* Kr    = alloc((size_t)4096 * 512);
  float* Vr    = alloc((size_t)4096 * 512);
  float* attnO = alloc((size_t)4096 * 512);
  float* tmp2  = alloc((size_t)4096 * 512);
  float* h1    = alloc((size_t)4096 * 2048);
  float* gathA = alloc((size_t)16384 * 64);
  float* tf    = alloc(1024);
  float* th1   = alloc(2048);
  float* te    = alloc(2048);

  auto gemm = [&](const float* A, const float* W, const float* bias, float* C,
                  int M, int N, int Kd, int act, int outmode, int bidx) {
    dim3 g((N + 127) / 128, M / 128);
    gemm_wmma<<<g, 256, 0, stream>>>(A, W, bias, C, M, N, Kd, act, outmode, bidx);
  };

  // ---- time embedding ----
  k_tfeat<<<4, 256, 0, stream>>>(tt, freqs, phases, tf);
  k_dense_small<<<8, 256, 0, stream>>>(tf, t_w1, t_b1, th1, 4, 256, 512, 1);
  k_dense_small<<<8, 256, 0, stream>>>(th1, t_w2, t_b2, te, 4, 512, 512, 0);

  // ---- input projections ----
  k_gather_in<<<4096, 256, 0, stream>>>(x, gathA);
  gemm(gathA, in_x_w, in_x_b, xh, 16384, 512, 64, 0, 0, 0);
  k_gather_in<<<4096, 256, 0, stream>>>(c, gathA);
  gemm(gathA, in_c_w, in_c_b, ch, 16384, 512, 64, 0, 0, 0);

  // ---- time stage setup ----
  k_ksum<<<8192, 256, 0, stream>>>(xh, x_cur);
  k_cond_time<<<8192, 256, 0, stream>>>(ch, te, sact);

  auto run_block = [&](float* xc, const BlockP& P, bool isTime) {
    gemm(sact, P.aw, P.ab, mods, 4096, 3072, 512, 0, 0, 0);
    k_lnmod<<<4096, 256, 0, stream>>>(xc, mods, 3072, 0, 512, tok);
    gemm(tok, P.qkv_w, P.qkv_b, qkv, 4096, 1536, 512, 0, 0, 0);
    if (isTime) {
      k_rope<<<4096, 256, 0, stream>>>(qkv, Qr, Kr, Vr);
      flash_attn<<<dim3(8, 8, 4), 256, 0, stream>>>(Qr, Kr, Vr, attnO);
    } else {
      dep_attn<<<1024, 256, 0, stream>>>(qkv, attnO);
    }
    gemm(attnO, P.proj_w, P.proj_b, tmp2, 4096, 512, 512, 0, 0, 0);
    k_resid<<<8192, 256, 0, stream>>>(xc, mods, 1024, tmp2);
    k_lnmod<<<4096, 256, 0, stream>>>(xc, mods, 3072, 1536, 2048, tok);
    gemm(tok, P.m1w, P.m1b, h1, 4096, 2048, 512, 1, 0, 0);
    gemm(h1, P.m2w, P.m2b, tmp2, 4096, 512, 2048, 0, 0, 0);
    k_resid<<<8192, 256, 0, stream>>>(xc, mods, 2560, tmp2);
  };

  // ---- time stage blocks ----
  for (int i = 0; i < 4; ++i) run_block(x_cur, tb[i], true);

  // ---- build dep-stage activations (x_cur now holds h) ----
  k_build_dep<<<32768, 256, 0, stream>>>(xh, x_cur, depX);

  // ---- dep stage, chunked per batch ----
  for (int b = 0; b < 4; ++b) {
    k_cond_dep<<<8192, 256, 0, stream>>>(ch, te, sact, b, 0);
    float* xc = depX + (size_t)b * 4096 * 512;
    for (int i = 0; i < 4; ++i) run_block(xc, db[i], false);
  }

  // ---- final layernorm-modulate + output projection ----
  for (int b = 0; b < 4; ++b) {
    k_gather_final<<<8192, 256, 0, stream>>>(depX, x_cur, b);
    k_cond_dep<<<8192, 256, 0, stream>>>(ch, te, sact, b, 1);
    gemm(sact, out_ada_w, out_ada_b, mods, 4096, 1024, 512, 0, 0, 0);
    k_lnmod<<<4096, 256, 0, stream>>>(x_cur, mods, 1024, 0, 512, tok);
    gemm(tok, out_w, out_b, (float*)d_out, 4096, 64, 512, 0, 1, b);
  }
}
